// LeechGPT_2336462209664
// MI455X (gfx1250) — matrix operand order repaired and checked by reading, not compiled, not run on hardware
//
#include <hip/hip_runtime.h>
#include <math.h>

// ---------------- problem constants ----------------
#define BC    4
#define TC    512
#define DC    768
#define LC    12
#define HC    8
#define HDC   96
#define E3    2304          // 3*D
#define F4    3072          // 4*D
#define VC    50257
#define MROWS (BC*TC)       // 2048 token rows
#define LN_EPS 1e-5f

// ---------------- WMMA types ----------------
typedef __attribute__((ext_vector_type(16))) __bf16 v16bf;
typedef __attribute__((ext_vector_type(8)))  float  v8f;

__device__ __forceinline__ __bf16 f32_to_bf16(float f) {
  union { float f; unsigned u; } a; a.f = f;
  unsigned r = a.u + 0x7FFFu + ((a.u >> 16) & 1u);   // round-to-nearest-even
  union { unsigned short s; __bf16 b; } c;
  c.s = (unsigned short)(r >> 16);
  return c.b;
}

__device__ __forceinline__ float gelu_exact(float x) {
  return 0.5f * x * (1.0f + erff(x * 0.70710678118654752f));
}

// =====================================================================
// Generic batched GEMM:  C[m,n] = act( alpha * sum_k A[m,k]*B[n,k] + bias[n] ) + resid[m,n]
// A: [M,K] bf16 row-major, B: [N,K] bf16 row-major (i.e. C = A * B^T)
// Block tile 128(M) x 128(N), K-step 32, double-buffered LDS.
// 8 waves in 4(M) x 2(N); wave tile 32x64 -> eight 16x16x32 bf16 WMMA
// per K step from 2 A-frags x 4 B-frags (1.5 ds_load_b128 per WMMA).
// =====================================================================
#define TM 128
#define TN 128
#define TK 32
#define LDS_ST 40   // row stride in bf16 elements (32 + 8 pad) -> 80B, 16B aligned

union FragU { v16bf v; uint4 u[2]; };

#define GF_GELU 1

__global__ __launch_bounds__(256) void gemm_bf16_kernel(
    const __bf16* __restrict__ A, long long sA,
    const __bf16* __restrict__ Bw, long long sB,
    float* __restrict__ C, long long sC, int ldc,
    __bf16* __restrict__ Cb, long long sCb,
    const float* __restrict__ bias,
    const float* __restrict__ resid, long long sR,
    int M, int N, int K, float alpha, int flags)
{
  __shared__ __bf16 As[2][TM * LDS_ST];   // 2 x 10 KB
  __shared__ __bf16 Bs[2][TN * LDS_ST];   // 2 x 10 KB

  const int z = blockIdx.z;
  A  += (long long)z * sA;
  Bw += (long long)z * sB;
  if (C)     C     += (long long)z * sC;
  if (Cb)    Cb    += (long long)z * sCb;
  if (resid) resid += (long long)z * sR;

  const int tid   = threadIdx.x;
  const int wave  = tid >> 5;
  const int lane  = tid & 31;
  const int waveM = wave & 3;     // 4 waves along M
  const int waveN = wave >> 2;    // 2 waves along N
  const int hl    = lane >> 4;    // half-wave select
  const int l16   = lane & 15;

  const int m0 = blockIdx.y * TM + waveM * 32;
  const int n0 = blockIdx.x * TN + waveN * 64;

  v8f acc[2][4] = {};

  // cooperative tile-load indices: 128 rows x 32 cols, 16 bf16 per thread
  const int row  = tid >> 1;           // 0..127
  const int colo = (tid & 1) * 16;     // 0 or 16
  const int gm = blockIdx.y * TM + row;     // M is always a multiple of TM here
  const int gn = blockIdx.x * TN + row;     // guarded against N

  const int nk = K / TK;

  // ---- prologue: stage K-tile 0 into buffer 0 ----
  {
    const uint4* ap = (const uint4*)(A + (long long)gm * K + colo);
    uint4 av0 = ap[0], av1 = ap[1];
    uint4* asp = (uint4*)&As[0][row * LDS_ST + colo];
    asp[0] = av0; asp[1] = av1;

    uint4 bv0 = make_uint4(0u, 0u, 0u, 0u), bv1 = bv0;
    if (gn < N) {
      const uint4* bp = (const uint4*)(Bw + (long long)gn * K + colo);
      bv0 = bp[0]; bv1 = bp[1];
    }
    uint4* bsp = (uint4*)&Bs[0][row * LDS_ST + colo];
    bsp[0] = bv0; bsp[1] = bv1;
  }
  __syncthreads();

  for (int kt = 0; kt < nk; ++kt) {
    const int cur = kt & 1;
    const int nxt = cur ^ 1;
    const bool has_next = (kt + 1) < nk;

    // ---- fetch next K-tile into registers (overlaps with WMMA below) ----
    uint4 av0, av1, bv0, bv1;
    if (has_next) {
      const int k0n = (kt + 1) * TK;
      const uint4* ap = (const uint4*)(A + (long long)gm * K + k0n + colo);
      av0 = ap[0]; av1 = ap[1];
      bv0 = make_uint4(0u, 0u, 0u, 0u); bv1 = bv0;
      if (gn < N) {
        const uint4* bp = (const uint4*)(Bw + (long long)gn * K + k0n + colo);
        bv0 = bp[0]; bv1 = bp[1];
        // pull the K-tile after that toward L2 (global_prefetch_b8)
        __builtin_prefetch(Bw + (long long)gn * K + k0n + TK + colo, 0, 1);
      }
    }

    // ---- fragments per ISA bf16 A(16x32)/B(32x16) lane layouts ----
    FragU af[2], bf[4];
    {
      const int ar = (waveM * 32 + l16) * LDS_ST + hl * 8;
      af[0].u[0] = *(const uint4*)&As[cur][ar];
      af[0].u[1] = *(const uint4*)&As[cur][ar + 16];
      af[1].u[0] = *(const uint4*)&As[cur][ar + 16 * LDS_ST];
      af[1].u[1] = *(const uint4*)&As[cur][ar + 16 * LDS_ST + 16];

      const int br = (waveN * 64 + l16) * LDS_ST + hl * 8;
#pragma unroll
      for (int j = 0; j < 4; ++j) {
        bf[j].u[0] = *(const uint4*)&Bs[cur][br + j * 16 * LDS_ST];
        bf[j].u[1] = *(const uint4*)&Bs[cur][br + j * 16 * LDS_ST + 16];
      }
    }

#pragma unroll
    for (int i = 0; i < 2; ++i)
#pragma unroll
      for (int j = 0; j < 4; ++j)
        acc[i][j] = __builtin_amdgcn_wmma_f32_16x16x32_bf16(
            false, af[i].v, false, bf[j].v, (short)0, acc[i][j], false, false);

    // ---- publish next tile into the other buffer ----
    if (has_next) {
      uint4* asp = (uint4*)&As[nxt][row * LDS_ST + colo];
      asp[0] = av0; asp[1] = av1;
      uint4* bsp = (uint4*)&Bs[nxt][row * LDS_ST + colo];
      bsp[0] = bv0; bsp[1] = bv1;
    }
    __syncthreads();
  }

  // ---- epilogue: scale, bias, gelu, residual; f32 and/or bf16 stores ----
#pragma unroll
  for (int i = 0; i < 2; ++i)
#pragma unroll
    for (int j = 0; j < 4; ++j)
#pragma unroll
      for (int r = 0; r < 8; ++r) {
        const int m = m0 + i * 16 + hl * 8 + r;
        const int n = n0 + j * 16 + l16;
        if (n < N) {
          float v = acc[i][j][r] * alpha;
          if (bias)            v += bias[n];
          if (flags & GF_GELU) v = gelu_exact(v);
          if (resid)           v += resid[(long long)m * ldc + n];
          const long long o = (long long)m * ldc + n;
          if (C)  C[o]  = v;
          if (Cb) Cb[o] = f32_to_bf16(v);
        }
      }
}

// =====================================================================
// 24x24 QR of the fixed Leech base matrix (modified Gram-Schmidt).
// =====================================================================
__global__ void leech_qr_kernel(float* __restrict__ kern) {
  if (threadIdx.x != 0 || blockIdx.x != 0) return;
  float A[24][24];
  float Q[24][24];
  for (int i = 0; i < 24; ++i)
    for (int j = 0; j < 24; ++j) A[i][j] = 0.f;
  for (int i = 0; i < 23; ++i) { A[i][i] = 2.f; A[i][i + 1] = 2.f; }
  A[23][23] = 2.f; A[23][0] = -2.f;
  for (int j = 0; j < 24; ++j) {
    float v[24];
    for (int i = 0; i < 24; ++i) v[i] = A[i][j];
    for (int p = 0; p < j; ++p) {
      float r = 0.f;
      for (int i = 0; i < 24; ++i) r += Q[i][p] * v[i];
      for (int i = 0; i < 24; ++i) v[i] -= r * Q[i][p];
    }
    float nrm = 0.f;
    for (int i = 0; i < 24; ++i) nrm += v[i] * v[i];
    const float inv = rsqrtf(nrm + 1e-30f);
    for (int i = 0; i < 24; ++i) Q[i][j] = v[i] * inv;
  }
  for (int i = 0; i < 24; ++i)
    for (int j = 0; j < 24; ++j) kern[i * 24 + j] = Q[i][j];
}

// =====================================================================
// Embedding gather: x[b,t,:] = tok_emb[idx[b,t],:] + pos_emb[t,:]
// =====================================================================
__global__ __launch_bounds__(256) void embed_kernel(
    const int* __restrict__ idx, const float* __restrict__ tok,
    const float* __restrict__ pos, float* __restrict__ X)
{
  const int bt = blockIdx.x;
  const int t = bt % TC;
  const int token = idx[bt];
  const float* tr = tok + (long long)token * DC;
  const float* pr = pos + (long long)t * DC;
  float* xr = X + (long long)bt * DC;
  for (int d = threadIdx.x; d < DC; d += 256) xr[d] = tr[d] + pr[d];
}

// =====================================================================
// LayerNorm (block per token row), optional f32 + bf16 outputs.
// =====================================================================
__global__ __launch_bounds__(256) void layernorm_kernel(
    const float* __restrict__ X, const float* __restrict__ gam,
    const float* __restrict__ bet, float* __restrict__ outf,
    __bf16* __restrict__ outb)
{
  __shared__ float sd[256];
  const long long row = blockIdx.x;
  const float* xr = X + row * DC;
  const int tid = threadIdx.x;

  float sum = 0.f;
  for (int d = tid; d < DC; d += 256) sum += xr[d];
  sd[tid] = sum; __syncthreads();
  for (int st = 128; st > 0; st >>= 1) { if (tid < st) sd[tid] += sd[tid + st]; __syncthreads(); }
  const float mean = sd[0] / (float)DC; __syncthreads();

  float vs = 0.f;
  for (int d = tid; d < DC; d += 256) { const float t = xr[d] - mean; vs += t * t; }
  sd[tid] = vs; __syncthreads();
  for (int st = 128; st > 0; st >>= 1) { if (tid < st) sd[tid] += sd[tid + st]; __syncthreads(); }
  const float rstd = rsqrtf(sd[0] / (float)DC + LN_EPS);

  for (int d = tid; d < DC; d += 256) {
    const float v = (xr[d] - mean) * rstd * gam[d] + bet[d];
    if (outf) outf[row * DC + d] = v;
    if (outb) outb[row * DC + d] = f32_to_bf16(v);
  }
}

// =====================================================================
// QKV split + per-24-block Leech rotation of Q and K.
// qkv row layout: e = c*768 + h*96 + hd   (c in {q,k,v})
// Outputs: Qr,Kr as [B*H, T, HD] bf16 ; Vt as [B*H, HD, T] bf16 (transposed)
// =====================================================================
__global__ __launch_bounds__(256) void split_rotate_kernel(
    const float* __restrict__ qkv, const float* __restrict__ kern,
    __bf16* __restrict__ Qr, __bf16* __restrict__ Kr, __bf16* __restrict__ Vt)
{
  __shared__ float kn[576];
  __shared__ float qbuf[DC];
  __shared__ float kbuf[DC];
  const int bt = blockIdx.x;          // 0..MROWS-1
  const int b = bt / TC;
  const int t = bt % TC;
  const float* row = qkv + (long long)bt * E3;

  for (int i = threadIdx.x; i < 576; i += 256) kn[i] = kern[i];
  for (int i = threadIdx.x; i < DC; i += 256) { qbuf[i] = row[i]; kbuf[i] = row[DC + i]; }
  __syncthreads();

  for (int i = threadIdx.x; i < DC; i += 256) {
    const int base = (i / 24) * 24;   // h*96 + blk*24
    const int j = i % 24;
    float aq = 0.f, ak = 0.f;
    for (int ii = 0; ii < 24; ++ii) {
      const float kv = kn[ii * 24 + j];
      aq += qbuf[base + ii] * kv;
      ak += kbuf[base + ii] * kv;
    }
    const int h = i / HDC;
    const int hd = i % HDC;
    const long long bh = (long long)(b * HC + h);
    const long long off = (bh * TC + t) * HDC + hd;
    Qr[off] = f32_to_bf16(aq);
    Kr[off] = f32_to_bf16(ak);
    const long long voff = (bh * HDC + hd) * TC + t;
    Vt[voff] = f32_to_bf16(row[2 * DC + i]);
  }
}

// =====================================================================
// Causal softmax over score rows; writes normalized probs in bf16.
// grid.x = B*H*T (one block per query row)
// =====================================================================
__global__ __launch_bounds__(256) void softmax_kernel(
    const float* __restrict__ S, __bf16* __restrict__ P)
{
  __shared__ float sd[256];
  const long long row = blockIdx.x;           // bh*T + q
  const int q = (int)(row % TC);
  const float* s = S + row * TC;
  __bf16* p = P + row * TC;
  const int tid = threadIdx.x;

  float mx = -3.4e38f;
  for (int j = tid; j < TC; j += 256) if (j <= q) mx = fmaxf(mx, s[j]);
  sd[tid] = mx; __syncthreads();
  for (int st = 128; st > 0; st >>= 1) { if (tid < st) sd[tid] = fmaxf(sd[tid], sd[tid + st]); __syncthreads(); }
  mx = sd[0]; __syncthreads();

  float sum = 0.f;
  for (int j = tid; j < TC; j += 256) if (j <= q) sum += __expf(s[j] - mx);
  sd[tid] = sum; __syncthreads();
  for (int st = 128; st > 0; st >>= 1) { if (tid < st) sd[tid] += sd[tid + st]; __syncthreads(); }
  const float inv = 1.f / sd[0];

  for (int j = tid; j < TC; j += 256) {
    const float v = (j <= q) ? __expf(s[j] - mx) * inv : 0.f;
    p[j] = f32_to_bf16(v);
  }
}

// =====================================================================
// Merge heads: o[B,H,T,HD] f32 -> obf[B,T,D] bf16
// =====================================================================
__global__ __launch_bounds__(256) void merge_heads_kernel(
    const float* __restrict__ O, __bf16* __restrict__ obf)
{
  long long i = (long long)blockIdx.x * 256 + threadIdx.x;
  const long long n = (long long)MROWS * DC;
  const long long stride = (long long)gridDim.x * 256;
  for (; i < n; i += stride) {
    const int d = (int)(i % DC);
    const long long bt = i / DC;
    const int t = (int)(bt % TC);
    const int b = (int)(bt / TC);
    const int h = d / HDC;
    const int hd = d % HDC;
    const long long src = (((long long)(b * HC + h)) * TC + t) * HDC + hd;
    obf[i] = f32_to_bf16(O[src]);
  }
}

// =====================================================================
// f32 -> bf16 elementwise (weight staging)
// =====================================================================
__global__ __launch_bounds__(256) void cvt_bf16_kernel(
    const float* __restrict__ in, __bf16* __restrict__ outp, long long n)
{
  long long i = (long long)blockIdx.x * 256 + threadIdx.x;
  const long long stride = (long long)gridDim.x * 256;
  for (; i < n; i += stride) outp[i] = f32_to_bf16(in[i]);
}

// =====================================================================
// Host orchestration
// =====================================================================
extern "C" void kernel_launch(void* const* d_in, const int* in_sizes, int n_in,
                              void* d_out, int out_size, void* d_ws, size_t ws_size,
                              hipStream_t stream) {
  (void)in_sizes; (void)n_in; (void)out_size; (void)ws_size;

  const int*   idx     = (const int*)  d_in[0];
  const float* tok_emb = (const float*)d_in[1];
  const float* pos_emb = (const float*)d_in[2];
  const float* ln1_s   = (const float*)d_in[3];
  const float* ln1_b   = (const float*)d_in[4];
  const float* qkv_w   = (const float*)d_in[5];
  const float* out_w   = (const float*)d_in[6];
  const float* ln2_s   = (const float*)d_in[7];
  const float* ln2_b   = (const float*)d_in[8];
  const float* ffn_w1  = (const float*)d_in[9];
  const float* ffn_b1  = (const float*)d_in[10];
  const float* ffn_w2  = (const float*)d_in[11];
  const float* ffn_b2  = (const float*)d_in[12];
  const float* fn_s    = (const float*)d_in[13];
  const float* fn_b    = (const float*)d_in[14];
  const float* head_w  = (const float*)d_in[15];

  float* logits = (float*)d_out;                     // [B,T,V]
  float* hidden = logits + (size_t)MROWS * VC;       // [B,T,D]

  // ---- workspace carve-out ----
  char* wsb = (char*)d_ws;
  size_t off = 0;
  auto alloc = [&](size_t bytes) -> void* {
    void* p = wsb + off;
    off = (off + bytes + 255) & ~(size_t)255;
    return p;
  };
  float*  kern  = (float*) alloc(576 * 4);
  float*  x     = (float*) alloc((size_t)MROWS * DC * 4);
  __bf16* hbf   = (__bf16*)alloc((size_t)MROWS * DC * 2);
  float*  qkv   = (float*) alloc((size_t)MROWS * E3 * 4);
  __bf16* Qr    = (__bf16*)alloc((size_t)MROWS * DC * 2);
  __bf16* Kr    = (__bf16*)alloc((size_t)MROWS * DC * 2);
  __bf16* Vt    = (__bf16*)alloc((size_t)MROWS * DC * 2);
  float*  sc    = (float*) alloc((size_t)BC * HC * TC * TC * 4);
  __bf16* Pm    = (__bf16*)alloc((size_t)BC * HC * TC * TC * 2);
  float*  oatt  = (float*) alloc((size_t)MROWS * DC * 4);   // [B,H,T,HD]
  __bf16* obf   = (__bf16*)alloc((size_t)MROWS * DC * 2);
  __bf16* fbf   = (__bf16*)alloc((size_t)MROWS * F4 * 2);
  __bf16* hidbf = (__bf16*)alloc((size_t)MROWS * DC * 2);
  __bf16* wbuf  = (__bf16*)alloc((size_t)VC * DC * 2);      // reused weight staging

  const dim3 blk(256);
  const float SCALE = 1.0f / sqrtf((float)HDC);

  auto cvt = [&](const float* src, __bf16* dst, long long n) {
    int blocks = (int)((n + 255) / 256);
    cvt_bf16_kernel<<<dim3(blocks), blk, 0, stream>>>(src, dst, n);
  };
  auto gemm = [&](const __bf16* A, long long sA, const __bf16* Bm, long long sB,
                  float* C, long long sC, int ldc, __bf16* Cb, long long sCb,
                  const float* bias, const float* resid, long long sR,
                  int M, int N, int K, float alpha, int flags, int batch) {
    dim3 g((N + TN - 1) / TN, M / TM, batch);
    gemm_bf16_kernel<<<g, blk, 0, stream>>>(A, sA, Bm, sB, C, sC, ldc, Cb, sCb,
                                            bias, resid, sR, M, N, K, alpha, flags);
  };

  // ---- constants + embedding ----
  leech_qr_kernel<<<dim3(1), dim3(1), 0, stream>>>(kern);
  embed_kernel<<<dim3(MROWS), blk, 0, stream>>>(idx, tok_emb, pos_emb, x);

  const long long sBH_qk = (long long)TC * HDC;   // Qr/Kr batch stride
  const long long sBH_v  = (long long)HDC * TC;   // Vt batch stride
  const long long sBH_s  = (long long)TC * TC;    // scores/P batch stride

  for (int l = 0; l < LC; ++l) {
    // LN1 -> bf16 activations
    layernorm_kernel<<<dim3(MROWS), blk, 0, stream>>>(x, ln1_s + l * DC, ln1_b + l * DC,
                                                      (float*)nullptr, hbf);
    // QKV projection
    cvt(qkv_w + (size_t)l * E3 * DC, wbuf, (long long)E3 * DC);
    gemm(hbf, 0, wbuf, 0, qkv, 0, E3, nullptr, 0, nullptr, nullptr, 0,
         MROWS, E3, DC, 1.0f, 0, 1);
    // split + Leech rotate
    split_rotate_kernel<<<dim3(MROWS), blk, 0, stream>>>(qkv, kern, Qr, Kr, Vt);
    // scores = SCALE * Qr x Kr^T  (batched over B*H)
    gemm(Qr, sBH_qk, Kr, sBH_qk, sc, sBH_s, TC, nullptr, 0, nullptr, nullptr, 0,
         TC, TC, HDC, SCALE, 0, BC * HC);
    // causal softmax -> P (bf16)
    softmax_kernel<<<dim3(BC * HC * TC), blk, 0, stream>>>(sc, Pm);
    // o = P x Vt^T  (batched)
    gemm(Pm, sBH_s, Vt, sBH_v, oatt, sBH_qk, HDC, nullptr, 0, nullptr, nullptr, 0,
         TC, HDC, TC, 1.0f, 0, BC * HC);
    // merge heads -> bf16 [B,T,D]
    merge_heads_kernel<<<dim3((MROWS * DC + 255) / 256), blk, 0, stream>>>(oatt, obf);
    // out projection + residual
    cvt(out_w + (size_t)l * DC * DC, wbuf, (long long)DC * DC);
    gemm(obf, 0, wbuf, 0, x, 0, DC, nullptr, 0, nullptr, x, 0,
         MROWS, DC, DC, 1.0f, 0, 1);
    // LN2
    layernorm_kernel<<<dim3(MROWS), blk, 0, stream>>>(x, ln2_s + l * DC, ln2_b + l * DC,
                                                      (float*)nullptr, hbf);
    // FFN1 (+bias, exact GELU) -> bf16
    cvt(ffn_w1 + (size_t)l * F4 * DC, wbuf, (long long)F4 * DC);
    gemm(hbf, 0, wbuf, 0, nullptr, 0, F4, fbf, 0, ffn_b1 + (size_t)l * F4,
         nullptr, 0, MROWS, F4, DC, 1.0f, GF_GELU, 1);
    // FFN2 (+bias, +residual)
    cvt(ffn_w2 + (size_t)l * DC * F4, wbuf, (long long)DC * F4);
    gemm(fbf, 0, wbuf, 0, x, 0, DC, nullptr, 0, ffn_b2 + (size_t)l * DC,
         x, 0, MROWS, DC, F4, 1.0f, 0, 1);
  }

  // final LN -> hidden (f32 to d_out) + bf16 for the head
  layernorm_kernel<<<dim3(MROWS), blk, 0, stream>>>(x, fn_s, fn_b, hidden, hidbf);

  // logits = hidden x head_w^T
  cvt(head_w, wbuf, (long long)VC * DC);
  gemm(hidbf, 0, wbuf, 0, logits, 0, VC, nullptr, 0, nullptr, nullptr, 0,
       MROWS, VC, DC, 1.0f, 0, 1);
}
